// CRFLayer_70162585747866
// MI455X (gfx1250) — matrix operand order repaired and checked by reading, not compiled, not run on hardware
//
#include <hip/hip_runtime.h>
#include <hip/hip_bf16.h>
#include <math.h>

// Problem constants (from reference): B=128, T=512, D=1024, K=32
#define BB   128
#define TT   512
#define DD   1024
#define KK   32
#define MROWS (BB * TT)          // 65536 GEMM rows
#define MTILES (MROWS / 16)      // 4096 16-row tiles

// Workspace layout
#define EM_BYTES   ((size_t)MROWS * KK * 4)                 // 8 MB emissions f32
#define HIST_OFF   (EM_BYTES)
#define HIST_BYTES ((size_t)(TT - 1) * BB * KK)             // ~2 MB u8 viterbi history
#define LL_OFF     (HIST_OFF + ((HIST_BYTES + 255) & ~(size_t)255))
#define LAST_OFF   (LL_OFF + 512)

typedef __attribute__((ext_vector_type(16))) __bf16          v16bf;
typedef __attribute__((ext_vector_type(8)))  float           v8f;
typedef __attribute__((ext_vector_type(16))) unsigned short  v16u;

// ---------------------------------------------------------------------------
// Kernel 1: emissions[M,32] = X[M,1024] @ W[1024,32] + b   (HBM-bound GEMM)
// One wave per 16-row tile; two 16x16 f32 accumulators cover N=32.
// W staged transposed in LDS as bf16 so each B-fragment is one 32B ds load.
// A fragments: native f32->bf16 casts of two 32B global vector loads.
// ---------------------------------------------------------------------------
__global__ void __launch_bounds__(256)
emis_gemm_wmma(const float* __restrict__ X, const float* __restrict__ W,
               const float* __restrict__ bias, float* __restrict__ em)
{
    __shared__ unsigned short Wlds[KK * DD];   // Wt[n][d], bf16: 64 KB (<320 KB/WGP)

    const int tid = threadIdx.x;
    for (int i = tid; i < DD * KK; i += 256) {       // W is L2-resident: cheap reload
        const int d = i >> 5, n = i & 31;            // W row-major [d][n]
        __bf16 w = (__bf16)W[i];
        Wlds[n * DD + d] = __builtin_bit_cast(unsigned short, w);
    }
    __syncthreads();

    const int lane = tid & 31;
    const int wid  = tid >> 5;
    const int m_l  = lane & 15;       // row within tile (A) / col within tile (B,C)
    const int h    = lane >> 4;       // lane half
    const int nwaves = (gridDim.x * blockDim.x) >> 5;

    for (int tile = blockIdx.x * 8 + wid; tile < MTILES; tile += nwaves) {
        const int row0 = tile * 16;
        // 16-bit A layout: lane half h holds K = 16h + e, e=0..15 (matches B)
        const float* __restrict__ xp = X + (size_t)(row0 + m_l) * DD + 16 * h;
        // 16-bit B layout used here (32x16): lane half h holds K = 16h + e, e=0..15
        const unsigned short* __restrict__ wp0 = &Wlds[m_l * DD + 16 * h];
        const unsigned short* __restrict__ wp1 = &Wlds[(m_l + 16) * DD + 16 * h];

        v8f c0 = {}; v8f c1 = {};

        #pragma unroll 2
        for (int kk = 0; kk < DD / 32; ++kk) {
            const int d0 = kk * 32;
            v8f x0 = *(const v8f*)(xp + d0);           // 32B, 32B-aligned
            v8f x1 = *(const v8f*)(xp + d0 + 8);
            __builtin_prefetch(xp + d0 + 64, 0, 3);    // global_prefetch_b8

            v16bf a;
            #pragma unroll
            for (int e = 0; e < 8; ++e) {              // native v_cvt f32->bf16
                a[e]     = (__bf16)x0[e];
                a[e + 8] = (__bf16)x1[e];
            }

            v16u bu0 = *(const v16u*)(wp0 + d0);       // one 32B LDS read each
            v16u bu1 = *(const v16u*)(wp1 + d0);
            v16bf b0 = __builtin_bit_cast(v16bf, bu0);
            v16bf b1 = __builtin_bit_cast(v16bf, bu1);

            c0 = __builtin_amdgcn_wmma_f32_16x16x32_bf16(false, a, false, b0,
                                                         (short)0, c0, false, false);
            c1 = __builtin_amdgcn_wmma_f32_16x16x32_bf16(false, a, false, b1,
                                                         (short)0, c1, false, false);
        }

        // C/D layout: VGPR r -> row r+8h; n = m_l (lanes 0-15) / m_l (16-31)
        const float bn0 = bias[m_l];
        const float bn1 = bias[m_l + 16];
        float* __restrict__ po = em + ((size_t)row0 + 8 * h) * KK + m_l;
        #pragma unroll
        for (int r = 0; r < 8; ++r) {                  // imm-offset store clause
            po[r * KK]      = c0[r] + bn0;
            po[r * KK + 16] = c1[r] + bn1;
        }
    }
}

// ---------------------------------------------------------------------------
// Kernel 2: per-batch CRF forward (logsumexp) + Viterbi forward + numerator.
// One wave per batch element; lane == tag (K=32 == wave32).
// ---------------------------------------------------------------------------
__global__ void __launch_bounds__(32)
crf_forward(const float* __restrict__ em, const int* __restrict__ labels,
            const unsigned char* __restrict__ mask,
            const float* __restrict__ startT, const float* __restrict__ endT,
            const float* __restrict__ trans,
            float* __restrict__ ll, int* __restrict__ lastTag,
            unsigned char* __restrict__ hist)
{
    const int b = blockIdx.x;
    const int lane = threadIdx.x;

    float tcol[KK];                       // column `lane` of trans, in VGPRs
    #pragma unroll
    for (int j = 0; j < KK; ++j) tcol[j] = trans[j * KK + lane];

    const float* __restrict__ eb = em + (size_t)b * TT * KK;

    float alpha = startT[lane] + eb[lane];   // forward log-potentials
    float vsc   = alpha;                     // viterbi scores

    for (int t = 1; t < TT; ++t) {
        const float e = eb[t * KK + lane];
        const int   m = mask[b * TT + t];

        float s[KK];
        float amax = -3.4e38f, vbest = -3.4e38f;
        int   jbest = 0;
        #pragma unroll
        for (int j = 0; j < KK; ++j) {
            const float aj = __shfl(alpha, j, 32);
            const float vj = __shfl(vsc,   j, 32);
            const float sj = aj + tcol[j];
            s[j] = sj;
            amax = fmaxf(amax, sj);
            const float svj = vj + tcol[j];
            if (svj > vbest) { vbest = svj; jbest = j; }   // first-max tie-break
        }
        float ssum = 0.f;
        #pragma unroll
        for (int j = 0; j < KK; ++j) ssum += __expf(s[j] - amax);

        hist[(size_t)(t - 1) * (BB * KK) + b * KK + lane] = (unsigned char)jbest;

        if (m) {
            alpha = amax + __logf(ssum) + e;
            vsc   = vbest + e;
        }
    }

    // denominator = logsumexp(alpha + end)
    float z = alpha + endT[lane];
    float zmax = z;
    #pragma unroll
    for (int off = 16; off; off >>= 1) zmax = fmaxf(zmax, __shfl_xor(zmax, off, 32));
    float ze = __expf(z - zmax);
    #pragma unroll
    for (int off = 16; off; off >>= 1) ze += __shfl_xor(ze, off, 32);
    const float denom = zmax + __logf(ze);

    // viterbi terminal argmax (lowest index wins ties, like jnp.argmax)
    float vz = vsc + endT[lane];
    int   vi = lane;
    #pragma unroll
    for (int off = 16; off; off >>= 1) {
        const float ov = __shfl_xor(vz, off, 32);
        const int   oi = __shfl_xor(vi, off, 32);
        if (ov > vz || (ov == vz && oi < vi)) { vz = ov; vi = oi; }
    }

    if (lane == 0) {
        // numerator: gold path score (sequential, trivial cost)
        const int* lb = labels + b * TT;
        const unsigned char* mb = mask + b * TT;
        int prev = lb[0];
        float sc = startT[prev] + eb[prev];
        for (int t = 1; t < TT; ++t) {
            const int tg = lb[t];
            const float step = trans[prev * KK + tg] + eb[t * KK + tg];
            if (mb[t]) { sc += step; prev = tg; }
        }
        sc += endT[prev];
        ll[b]      = sc - denom;
        lastTag[b] = vi;
    }
}

// ---------------------------------------------------------------------------
// Kernel 3: backtrack per batch + loss reduction. d_out[0]=loss, [1..]=tags.
// ---------------------------------------------------------------------------
__global__ void __launch_bounds__(128)
crf_backtrack(const float* __restrict__ ll, const int* __restrict__ lastTag,
              const unsigned char* __restrict__ hist, float* __restrict__ out)
{
    const int b = threadIdx.x;
    float* tags = out + 1;
    int tag = lastTag[b];
    tags[b * TT + (TT - 1)] = (float)tag;
    for (int t = TT - 2; t >= 0; --t) {
        tag = hist[(size_t)t * (BB * KK) + b * KK + tag];
        tags[b * TT + t] = (float)tag;
    }
    __syncthreads();
    if (b == 0) {
        float s = 0.f;
        for (int i = 0; i < BB; ++i) s += ll[i];
        out[0] = -s;                      // reduction='sum', module returns -ll
    }
}

extern "C" void kernel_launch(void* const* d_in, const int* in_sizes, int n_in,
                              void* d_out, int out_size, void* d_ws, size_t ws_size,
                              hipStream_t stream) {
    const float*         X      = (const float*)d_in[0];
    const int*           labels = (const int*)d_in[1];
    const unsigned char* masks  = (const unsigned char*)d_in[2];  // jnp bool = 1B
    const float*         W      = (const float*)d_in[3];
    const float*         bias   = (const float*)d_in[4];
    const float*         startT = (const float*)d_in[5];
    const float*         endT   = (const float*)d_in[6];
    const float*         trans  = (const float*)d_in[7];

    float*         em      = (float*)d_ws;
    unsigned char* hist    = (unsigned char*)d_ws + HIST_OFF;
    float*         ll      = (float*)((char*)d_ws + LL_OFF);
    int*           lastTag = (int*)((char*)d_ws + LAST_OFF);

    // 512 blocks x 8 waves = 4096 waves = one per 16-row tile
    emis_gemm_wmma<<<512, 256, 0, stream>>>(X, W, bias, em);
    crf_forward<<<BB, 32, 0, stream>>>(em, labels, masks, startT, endT, trans,
                                       ll, lastTag, hist);
    crf_backtrack<<<1, BB, 0, stream>>>(ll, lastTag, hist, (float*)d_out);
}
